// JointGNN_37529424232875
// MI455X (gfx1250) — compile-verified, matrix operand order checked by reading, not run on hardware
//
#include <hip/hip_runtime.h>
#include <hip/hip_bf16.h>

// ---------------- problem constants (fixed by the reference) ----------------
#define N_VAR   100000
#define N_CONS  50000
#define N_FEAS  150000   // N_VAR + N_CONS
#define E_OBJ   1600000
#define E_FEAS  2400000
#define HDIM    64       // GCN feature width
#define CDIM    128      // concat width
#define DDIM    256      // MLP hidden
#define NUM_GRAPHS 64
#define N_BIN   50000

typedef __attribute__((ext_vector_type(16))) __bf16 v16bf;
typedef __attribute__((ext_vector_type(8)))  float  v8f;
typedef __attribute__((ext_vector_type(4)))  float  f4v;

static inline unsigned cdiv_u(long n, long d) { return (unsigned)((n + d - 1) / d); }

// ---------------------------------------------------------------------------
// elementwise helpers
// ---------------------------------------------------------------------------
__global__ void fill_kernel(float* __restrict__ p, float v, long n) {
    long i = (long)blockIdx.x * blockDim.x + threadIdx.x;
    if (i < n) p[i] = v;
}

__global__ void deg_kernel(const int* __restrict__ dst, const float* __restrict__ ew,
                           float* __restrict__ deg, int E) {
    int e = blockIdx.x * blockDim.x + threadIdx.x;
    if (e >= E) return;
    atomicAdd(&deg[dst[e]], ew[e]);
}

__global__ void rsqrt_kernel(float* __restrict__ deg, int n) {
    int i = blockIdx.x * blockDim.x + threadIdx.x;
    if (i >= n) return;
    float d = deg[i];
    deg[i] = (d > 0.0f) ? rsqrtf(d) : 0.0f;
}

// layer-1 dense: out[i,j] = sum_k x[i,k(4)] * W[k,j(64)]
__global__ void xw1_kernel(const float* __restrict__ x, const float* __restrict__ W,
                           float* __restrict__ out, int N) {
    long tid = (long)blockIdx.x * blockDim.x + threadIdx.x;
    long i = tid >> 6; int j = (int)(tid & 63);
    if (i >= N) return;
    const float* xr = x + i * 4;
    float s = xr[0] * W[j] + xr[1] * W[64 + j] + xr[2] * W[128 + j] + xr[3] * W[192 + j];
    out[i * HDIM + j] = s;
}

// edge scatter: one wave (32 lanes) per edge, 2 channels per lane
__global__ void edge_scatter_kernel(const float* __restrict__ h,
                                    const int* __restrict__ src, const int* __restrict__ dst,
                                    const float* __restrict__ ew, const float* __restrict__ dinv,
                                    float* __restrict__ out, int E) {
    long tid = (long)blockIdx.x * blockDim.x + threadIdx.x;
    long e = tid >> 5;
    if (e >= E) return;
    int lane = threadIdx.x & 31;
    int s = src[e], d = dst[e];                 // uniform per wave (hw broadcast)
    float coef = dinv[s] * ew[e] * dinv[d];
    float2 hv = *(const float2*)(h + (long)s * HDIM + lane * 2);
    atomicAdd(out + (long)d * HDIM + lane * 2,     hv.x * coef);
    atomicAdd(out + (long)d * HDIM + lane * 2 + 1, hv.y * coef);
}

// out = relu(out + dinv[i]^2 * h + b)   (self-loop + bias + relu)
__global__ void finalize_kernel(float* __restrict__ out, const float* __restrict__ h,
                                const float* __restrict__ dinv, const float* __restrict__ b,
                                int N) {
    long tid = (long)blockIdx.x * blockDim.x + threadIdx.x;
    long i = tid >> 6; int c = (int)(tid & 63);
    if (i >= N) return;
    float di = dinv[i];
    float v = out[i * HDIM + c] + di * di * h[i * HDIM + c] + b[c];
    out[i * HDIM + c] = fmaxf(v, 0.0f);
}

// weight transpose + bf16 convert: Wt[m*K+k] = (bf16) W[k*M+m]
__global__ void cvt_transpose_kernel(const float* __restrict__ W, __bf16* __restrict__ Wt,
                                     int K, int M) {
    int idx = blockIdx.x * blockDim.x + threadIdx.x;
    if (idx >= K * M) return;
    int m = idx / K, k = idx - m * K;
    Wt[idx] = (__bf16)W[(long)k * M + m];
}

// z_shared assembly (150000 x 128)
__global__ void build_zshared_kernel(const float* __restrict__ zobj,
                                     const float* __restrict__ hfeas,
                                     float* __restrict__ zs) {
    long idx = (long)blockIdx.x * blockDim.x + threadIdx.x;
    if (idx >= (long)N_FEAS * CDIM) return;
    long row = idx >> 7; int c = (int)(idx & 127);
    float v;
    if (row < N_VAR) v = (c < 64) ? zobj[row * HDIM + c] : hfeas[row * HDIM + (c - 64)];
    else             v = (c < 64) ? hfeas[row * HDIM + c] : 0.0f;
    zs[idx] = v;
}

// reparameterization: z = mu + exp(0.5*lv) * eps,  eps ~ N(0,1) via hash Box-Muller
__device__ __forceinline__ unsigned pcg_hash(unsigned x) {
    x = x * 747796405u + 2891336453u;
    unsigned w = ((x >> ((x >> 28) + 4u)) ^ x) * 277803737u;
    return (w >> 22) ^ w;
}
__global__ void sample_kernel(const float* __restrict__ mu, const float* __restrict__ lv,
                              float* __restrict__ z, long n) {
    long i = (long)blockIdx.x * blockDim.x + threadIdx.x;
    if (i >= n) return;
    unsigned h1 = pcg_hash((unsigned)i);
    unsigned h2 = pcg_hash((unsigned)i ^ 0x9E3779B9u);
    float u1 = ((float)h1 + 1.0f) * 2.3283064e-10f;   // (0,1]
    float u2 = (float)h2 * 2.3283064e-10f;
    float eps = sqrtf(-2.0f * __logf(u1)) * __cosf(6.28318530718f * u2);
    z[i] = mu[i] + __expf(0.5f * lv[i]) * eps;
}

// segment-mean pooling
__global__ void pool_accum_kernel(const float* __restrict__ z, const int* __restrict__ batch,
                                  float* __restrict__ pooled, float* __restrict__ cnt, int nvar) {
    long tid = (long)blockIdx.x * blockDim.x + threadIdx.x;
    long row = tid >> 7; int c = (int)(tid & 127);
    if (row >= nvar) return;
    int g = batch[row];
    atomicAdd(&pooled[(long)g * CDIM + c], z[row * CDIM + c]);
    if (c == 0) atomicAdd(&cnt[g], 1.0f);
}
__global__ void pool_div_kernel(float* __restrict__ pooled, const float* __restrict__ cnt) {
    int tid = blockIdx.x * blockDim.x + threadIdx.x;
    if (tid >= NUM_GRAPHS * CDIM) return;
    pooled[tid] /= fmaxf(cnt[tid >> 7], 1.0f);
}

// ---------------------------------------------------------------------------
// WMMA GEMM:  Out[N,M] = act(A[N,K] @ W[K,M] + bias)
// Wt is bf16, pre-transposed: Wt[m*K + k].  K in {64,128}, M multiple of 16.
// 256 threads = 8 waves; 128 rows per block; A staged fp32->bf16 in LDS.
// ---------------------------------------------------------------------------
__global__ __launch_bounds__(256) void wmma_gemm_kernel(
    const float* __restrict__ A, const __bf16* __restrict__ Wt,
    const float* __restrict__ bias, float* __restrict__ Out,
    int N, int K, int M, int act_relu) {
    extern __shared__ __bf16 lds[];        // 128 x (K+8)
    const int ldk = K + 8;
    const int tid = threadIdx.x;
    const int wave = tid >> 5, lane = tid & 31;
    const int rowBase = blockIdx.x * 128;

    for (int idx = tid; idx < 128 * K; idx += 256) {
        int r = idx / K, k = idx - r * K;
        int row = rowBase + r;
        float v = (row < N) ? A[(long)row * K + k] : 0.0f;
        lds[r * ldk + k] = (__bf16)v;
    }
    __syncthreads();

    const int halfsel = lane >> 4;
    const int l16 = lane & 15;
    const int ko = halfsel * 8;
    const __bf16* arow = lds + (wave * 16 + l16) * ldk;
    const int nct = M >> 4, nkt = K >> 5;

    for (int ct = 0; ct < nct; ++ct) {
        v8f acc = {};
        const __bf16* bcol = Wt + (long)(ct * 16 + l16) * K;
        for (int kt = 0; kt < nkt; ++kt) {
            v16bf af, bf;
            const __bf16* ap = arow + kt * 32 + ko;
            ((f4v*)&af)[0] = *(const f4v*)(ap);
            ((f4v*)&af)[1] = *(const f4v*)(ap + 16);
            const __bf16* bp = bcol + kt * 32 + ko;
            ((f4v*)&bf)[0] = *(const f4v*)(bp);
            ((f4v*)&bf)[1] = *(const f4v*)(bp + 16);
            acc = __builtin_amdgcn_wmma_f32_16x16x32_bf16(false, af, false, bf,
                                                          (short)0, acc, false, false);
        }
        int colg = ct * 16 + l16;
        float bv = bias ? bias[colg] : 0.0f;
#pragma unroll
        for (int v = 0; v < 8; ++v) {
            int row = rowBase + wave * 16 + v + halfsel * 8;
            if (row < N) {
                float o = acc[v] + bv;
                if (act_relu) o = fmaxf(o, 0.0f);
                Out[(long)row * M + colg] = o;
            }
        }
    }
}

// ---------------------------------------------------------------------------
// Fused MLP head:  out[i] = maybe_sigmoid( relu(Z[i,:128] @ W1 + b1) @ W2 + b2 )
// W1t bf16 pre-transposed [256 x 128]; hidden never leaves registers.
// ---------------------------------------------------------------------------
__global__ __launch_bounds__(256) void mlp_head_kernel(
    const float* __restrict__ Z, const __bf16* __restrict__ W1t,
    const float* __restrict__ b1, const float* __restrict__ W2,
    const float* __restrict__ b2p, float* __restrict__ out,
    int N, int act_sigmoid) {
    __shared__ __bf16 lds[128 * (CDIM + 8)];
    const int ldk = CDIM + 8;
    const int tid = threadIdx.x;
    const int wave = tid >> 5, lane = tid & 31;
    const int rowBase = blockIdx.x * 128;

    for (int idx = tid; idx < 128 * CDIM; idx += 256) {
        int r = idx >> 7, k = idx & 127;
        int row = rowBase + r;
        float v = (row < N) ? Z[(long)row * CDIM + k] : 0.0f;
        lds[r * ldk + k] = (__bf16)v;
    }
    __syncthreads();

    const int halfsel = lane >> 4;
    const int l16 = lane & 15;
    const int ko = halfsel * 8;
    const __bf16* arow = lds + (wave * 16 + l16) * ldk;

    float partial[8] = {0, 0, 0, 0, 0, 0, 0, 0};
    for (int ht = 0; ht < DDIM / 16; ++ht) {
        v8f acc = {};
        int col = ht * 16 + l16;
        const __bf16* bcol = W1t + (long)col * CDIM;
#pragma unroll
        for (int kt = 0; kt < CDIM / 32; ++kt) {
            v16bf af, bf;
            const __bf16* ap = arow + kt * 32 + ko;
            ((f4v*)&af)[0] = *(const f4v*)(ap);
            ((f4v*)&af)[1] = *(const f4v*)(ap + 16);
            const __bf16* bp = bcol + kt * 32 + ko;
            ((f4v*)&bf)[0] = *(const f4v*)(bp);
            ((f4v*)&bf)[1] = *(const f4v*)(bp + 16);
            acc = __builtin_amdgcn_wmma_f32_16x16x32_bf16(false, af, false, bf,
                                                          (short)0, acc, false, false);
        }
        float bb = b1[col], w2 = W2[col];
#pragma unroll
        for (int v = 0; v < 8; ++v) {
            float hv = fmaxf(acc[v] + bb, 0.0f);
            partial[v] += hv * w2;
        }
    }
#pragma unroll
    for (int v = 0; v < 8; ++v) {
        float s = partial[v];
        s += __shfl_xor(s, 1, 32);
        s += __shfl_xor(s, 2, 32);
        s += __shfl_xor(s, 4, 32);
        s += __shfl_xor(s, 8, 32);
        partial[v] = s;
    }
    if (l16 == 0) {
        float b2 = b2p[0];
#pragma unroll
        for (int v = 0; v < 8; ++v) {
            int row = rowBase + wave * 16 + v + halfsel * 8;
            if (row < N) {
                float o = partial[v] + b2;
                if (act_sigmoid) o = 1.0f / (1.0f + __expf(-o));
                out[row] = o;
            }
        }
    }
}

// ---------------------------------------------------------------------------
extern "C" void kernel_launch(void* const* d_in, const int* in_sizes, int n_in,
                              void* d_out, int out_size, void* d_ws, size_t ws_size,
                              hipStream_t stream) {
    const float* x_obj   = (const float*)d_in[0];
    const int*   ei_obj  = (const int*)  d_in[1];
    const float* ew_obj  = (const float*)d_in[2];
    const float* x_feas  = (const float*)d_in[3];
    const int*   ei_feas = (const int*)  d_in[4];
    const float* ew_feas = (const float*)d_in[5];
    const int*   batch   = (const int*)  d_in[6];
    const float* c1o_W = (const float*)d_in[9],  *c1o_b = (const float*)d_in[10];
    const float* c2o_W = (const float*)d_in[11], *c2o_b = (const float*)d_in[12];
    const float* c1c_W = (const float*)d_in[13], *c1c_b = (const float*)d_in[14];
    const float* c2c_W = (const float*)d_in[15], *c2c_b = (const float*)d_in[16];
    const float* mu_W  = (const float*)d_in[17], *mu_b  = (const float*)d_in[18];
    const float* lv_W  = (const float*)d_in[19], *lv_b  = (const float*)d_in[20];
    const float* dx_W1 = (const float*)d_in[21], *dx_b1 = (const float*)d_in[22];
    const float* dx_W2 = (const float*)d_in[23], *dx_b2 = (const float*)d_in[24];
    const float* dc_W1 = (const float*)d_in[25], *dc_b1 = (const float*)d_in[26];
    const float* dc_W2 = (const float*)d_in[27], *dc_b2 = (const float*)d_in[28];
    const float* dk_W1 = (const float*)d_in[29], *dk_b1 = (const float*)d_in[30];
    const float* dk_W2 = (const float*)d_in[31], *dk_b2 = (const float*)d_in[32];
    const float* di_W1 = (const float*)d_in[33], *di_b1 = (const float*)d_in[34];
    const float* di_W2 = (const float*)d_in[35], *di_b2 = (const float*)d_in[36];

    float* out = (float*)d_out;
    float* out_xhat  = out;
    float* out_cost  = out + 100000;
    float* out_cons  = out + 100064;
    float* out_integ = out + 150064;
    float* out_zmu   = out + 200064;                 // 150000*128
    float* out_zlv   = out + 200064 + 19200000;

    // workspace layout (floats)
    float* ws = (float*)d_ws;
    float* deg_obj  = ws;                            // 100000
    float* deg_feas = ws + 100352;                   // 150000
    float* pooled   = ws + 250880;                   // 8192
    float* cnt      = pooled + NUM_GRAPHS * CDIM;    // 64
    __bf16* wb      = (__bf16*)(ws + 259136);        // 172032 bf16
    __bf16* wb_c2o = wb + 0;
    __bf16* wb_c2c = wb + 4096;
    __bf16* wb_mu  = wb + 8192;
    __bf16* wb_lv  = wb + 24576;
    __bf16* wb_dx1 = wb + 40960;
    __bf16* wb_dc1 = wb + 73728;
    __bf16* wb_dk1 = wb + 106496;
    __bf16* wb_di1 = wb + 139264;
    float* bufA = ws + 345600;                       // 150000*64
    float* bufB = bufA + 9600000;                    // 150000*64
    float* zobj = bufB + 9600000;                    // 100000*64
    float* zsh  = zobj + 6400000;                    // 150000*128 (reused as sampled z)

    const int TB = 256;
    size_t lds64  = (size_t)128 * (64 + 8) * sizeof(__bf16);
    size_t lds128 = (size_t)128 * (128 + 8) * sizeof(__bf16);

    // ---- weight prep (bf16, transposed) ----
    cvt_transpose_kernel<<<cdiv_u(64 * 64, TB), TB, 0, stream>>>(c2o_W, wb_c2o, 64, 64);
    cvt_transpose_kernel<<<cdiv_u(64 * 64, TB), TB, 0, stream>>>(c2c_W, wb_c2c, 64, 64);
    cvt_transpose_kernel<<<cdiv_u(128 * 128, TB), TB, 0, stream>>>(mu_W, wb_mu, 128, 128);
    cvt_transpose_kernel<<<cdiv_u(128 * 128, TB), TB, 0, stream>>>(lv_W, wb_lv, 128, 128);
    cvt_transpose_kernel<<<cdiv_u(128 * 256, TB), TB, 0, stream>>>(dx_W1, wb_dx1, 128, 256);
    cvt_transpose_kernel<<<cdiv_u(128 * 256, TB), TB, 0, stream>>>(dc_W1, wb_dc1, 128, 256);
    cvt_transpose_kernel<<<cdiv_u(128 * 256, TB), TB, 0, stream>>>(dk_W1, wb_dk1, 128, 256);
    cvt_transpose_kernel<<<cdiv_u(128 * 256, TB), TB, 0, stream>>>(di_W1, wb_di1, 128, 256);

    // ---- degrees / normalization ----
    fill_kernel<<<cdiv_u(N_VAR, TB), TB, 0, stream>>>(deg_obj, 1.0f, N_VAR);
    fill_kernel<<<cdiv_u(N_FEAS, TB), TB, 0, stream>>>(deg_feas, 1.0f, N_FEAS);
    deg_kernel<<<cdiv_u(E_OBJ, TB), TB, 0, stream>>>(ei_obj + E_OBJ, ew_obj, deg_obj, E_OBJ);
    deg_kernel<<<cdiv_u(E_FEAS, TB), TB, 0, stream>>>(ei_feas + E_FEAS, ew_feas, deg_feas, E_FEAS);
    rsqrt_kernel<<<cdiv_u(N_VAR, TB), TB, 0, stream>>>(deg_obj, N_VAR);
    rsqrt_kernel<<<cdiv_u(N_FEAS, TB), TB, 0, stream>>>(deg_feas, N_FEAS);

    // ---- obj GCN layer 1 ----
    xw1_kernel<<<cdiv_u((long)N_VAR * 64, TB), TB, 0, stream>>>(x_obj, c1o_W, bufA, N_VAR);
    hipMemsetAsync(bufB, 0, (size_t)N_VAR * 64 * sizeof(float), stream);
    edge_scatter_kernel<<<cdiv_u((long)E_OBJ * 32, TB), TB, 0, stream>>>(
        bufA, ei_obj, ei_obj + E_OBJ, ew_obj, deg_obj, bufB, E_OBJ);
    finalize_kernel<<<cdiv_u((long)N_VAR * 64, TB), TB, 0, stream>>>(bufB, bufA, deg_obj, c1o_b, N_VAR);

    // ---- obj GCN layer 2 ----
    wmma_gemm_kernel<<<cdiv_u(N_VAR, 128), TB, lds64, stream>>>(
        bufB, wb_c2o, nullptr, bufA, N_VAR, 64, 64, 0);
    hipMemsetAsync(zobj, 0, (size_t)N_VAR * 64 * sizeof(float), stream);
    edge_scatter_kernel<<<cdiv_u((long)E_OBJ * 32, TB), TB, 0, stream>>>(
        bufA, ei_obj, ei_obj + E_OBJ, ew_obj, deg_obj, zobj, E_OBJ);
    finalize_kernel<<<cdiv_u((long)N_VAR * 64, TB), TB, 0, stream>>>(zobj, bufA, deg_obj, c2o_b, N_VAR);

    // ---- feas GCN layer 1 ----
    xw1_kernel<<<cdiv_u((long)N_FEAS * 64, TB), TB, 0, stream>>>(x_feas, c1c_W, bufA, N_FEAS);
    hipMemsetAsync(bufB, 0, (size_t)N_FEAS * 64 * sizeof(float), stream);
    edge_scatter_kernel<<<cdiv_u((long)E_FEAS * 32, TB), TB, 0, stream>>>(
        bufA, ei_feas, ei_feas + E_FEAS, ew_feas, deg_feas, bufB, E_FEAS);
    finalize_kernel<<<cdiv_u((long)N_FEAS * 64, TB), TB, 0, stream>>>(bufB, bufA, deg_feas, c1c_b, N_FEAS);

    // ---- feas GCN layer 2 ----
    wmma_gemm_kernel<<<cdiv_u(N_FEAS, 128), TB, lds64, stream>>>(
        bufB, wb_c2c, nullptr, bufA, N_FEAS, 64, 64, 0);
    hipMemsetAsync(bufB, 0, (size_t)N_FEAS * 64 * sizeof(float), stream);
    edge_scatter_kernel<<<cdiv_u((long)E_FEAS * 32, TB), TB, 0, stream>>>(
        bufA, ei_feas, ei_feas + E_FEAS, ew_feas, deg_feas, bufB, E_FEAS);
    finalize_kernel<<<cdiv_u((long)N_FEAS * 64, TB), TB, 0, stream>>>(bufB, bufA, deg_feas, c2c_b, N_FEAS);

    // ---- z_shared (150000 x 128) ----
    build_zshared_kernel<<<cdiv_u((long)N_FEAS * CDIM, TB), TB, 0, stream>>>(zobj, bufB, zsh);

    // ---- z_mu / z_logvar (written directly to output) ----
    wmma_gemm_kernel<<<cdiv_u(N_FEAS, 128), TB, lds128, stream>>>(
        zsh, wb_mu, mu_b, out_zmu, N_FEAS, 128, 128, 0);
    wmma_gemm_kernel<<<cdiv_u(N_FEAS, 128), TB, lds128, stream>>>(
        zsh, wb_lv, lv_b, out_zlv, N_FEAS, 128, 128, 0);

    // ---- reparameterization sample (z overwrites z_shared) ----
    sample_kernel<<<cdiv_u((long)N_FEAS * CDIM, TB), TB, 0, stream>>>(
        out_zmu, out_zlv, zsh, (long)N_FEAS * CDIM);

    // ---- heads ----
    mlp_head_kernel<<<cdiv_u(N_VAR, 128), TB, 0, stream>>>(
        zsh, wb_dx1, dx_b1, dx_W2, dx_b2, out_xhat, N_VAR, 0);

    hipMemsetAsync(pooled, 0, (size_t)(NUM_GRAPHS * CDIM + NUM_GRAPHS) * sizeof(float), stream);
    pool_accum_kernel<<<cdiv_u((long)N_VAR * CDIM, TB), TB, 0, stream>>>(zsh, batch, pooled, cnt, N_VAR);
    pool_div_kernel<<<cdiv_u(NUM_GRAPHS * CDIM, TB), TB, 0, stream>>>(pooled, cnt);
    mlp_head_kernel<<<1, TB, 0, stream>>>(
        pooled, wb_dc1, dc_b1, dc_W2, dc_b2, out_cost, NUM_GRAPHS, 0);

    mlp_head_kernel<<<cdiv_u(N_CONS, 128), TB, 0, stream>>>(
        zsh + (long)N_VAR * CDIM, wb_dk1, dk_b1, dk_W2, dk_b2, out_cons, N_CONS, 0);

    mlp_head_kernel<<<cdiv_u(N_BIN, 128), TB, 0, stream>>>(
        zsh, wb_di1, di_b1, di_W2, di_b2, out_integ, N_BIN, 1);
}
